// RecencyPreferenceEnergy_90091234001027
// MI455X (gfx1250) — matrix operand order repaired and checked by reading, not compiled, not run on hardware
//
#include <hip/hip_runtime.h>
#include <hip/hip_bf16.h>
#include <math.h>

typedef __bf16 bf16_t;
typedef __attribute__((ext_vector_type(16))) __bf16 v16bf;
typedef __attribute__((ext_vector_type(8)))  __bf16 v8bf;
typedef __attribute__((ext_vector_type(8)))  float  v8f;

#define K_TOK   64
#define D_FEAT  384
#define H_DIM   64
#define NPAIR   28

// np.triu_indices(8, k=1) row-major order
__constant__ int c_ii[NPAIR] = {0,0,0,0,0,0,0, 1,1,1,1,1,1, 2,2,2,2,2, 3,3,3,3, 4,4,4, 5,5, 6};
__constant__ int c_jj[NPAIR] = {1,2,3,4,5,6,7, 2,3,4,5,6,7, 3,4,5,6,7, 4,5,6,7, 5,6,7, 6,7, 7};

__device__ __forceinline__ float silu_f(float x) {
    return x / (1.0f + __expf(-x));
}

__device__ __forceinline__ v16bf cat8(v8bf lo, v8bf hi) {
    return __builtin_shufflevector(lo, hi, 0,1,2,3,4,5,6,7,8,9,10,11,12,13,14,15);
}

// ---------------------------------------------------------------------------
// Prep: convert W1 (771x64) and W2 (64x32) to bf16 in workspace (L2-resident).
// ---------------------------------------------------------------------------
__global__ void pack_weights_bf16(const float* __restrict__ W1,
                                  const float* __restrict__ W2,
                                  bf16_t* __restrict__ W1bf,
                                  bf16_t* __restrict__ W2bf) {
    int i = blockIdx.x * 256 + threadIdx.x;
    if (i < 771 * 64) W1bf[i] = (bf16_t)W1[i];
    if (i < 64 * 32)  W2bf[i] = (bf16_t)W2[i];
}

// ---------------------------------------------------------------------------
// Fused kernel: one 256-thread workgroup (8 wave32) handles TWO batches.
//   1. in-wave top-8 argmax of y (stable ties, matches jax.lax.top_k)
//   2. gather 16 rows of M -> LDS bf16
//   3. G(16x128) = Mtop @ [W1a|W1b] via v_wmma_f32_16x16x32_bf16 (8 waves x 12 K)
//   4. assemble 2x28 pair rows: silu(G_i + G_j + dt*w768 + yi*w769 + yj*w770 + b1)
//   5. h2 tiles via WMMA (8 jobs, 1/wave), silu, dot W3, shuffle + ds_add_f32 reduce
//   6. out[b] = sum_p (pen_p + b3) * yi*yj
// ---------------------------------------------------------------------------
__global__ __launch_bounds__(256) void
recency_energy_kernel(const float* __restrict__ M,
                      const float* __restrict__ y,
                      const float* __restrict__ ts,
                      const float* __restrict__ W1,   // f32, rows 768..770 + b1 path
                      const float* __restrict__ b1,
                      const float* __restrict__ b2,
                      const float* __restrict__ W3,
                      const float* __restrict__ b3,
                      const bf16_t* __restrict__ W1bf,
                      const bf16_t* __restrict__ W2bf,
                      float* __restrict__ out) {
    __shared__ bf16_t sA[16][D_FEAT];      // 12 KB: 2 batches x 8 top rows, bf16
    __shared__ float  sG[16][128];         //  8 KB: layer-1 output
    __shared__ bf16_t sH[2][32][H_DIM];    //  8 KB: layer-2 A operand (28 rows + pad)
    __shared__ float  sPen[2][32];
    __shared__ int    sIdx[2][8];
    __shared__ float  sY[2][8];
    __shared__ float  sT[2][8];

    const int tid  = threadIdx.x;
    const int lane = tid & 31;
    const int wave = tid >> 5;
    const int b0   = blockIdx.x * 2;

    // ---- Step 1: top-8 (waves 0,1; one batch each; shuffle argmax, stable) ----
    if (wave < 2) {
        const long long ybase = (long long)(b0 + wave) * K_TOK;
        float v0 = y[ybase + 2 * lane];
        float v1 = y[ybase + 2 * lane + 1];
        #pragma unroll
        for (int r = 0; r < 8; ++r) {
            float bv; int bi;
            if (v0 >= v1) { bv = v0; bi = 2 * lane; }
            else          { bv = v1; bi = 2 * lane + 1; }
            #pragma unroll
            for (int off = 16; off >= 1; off >>= 1) {
                float ov = __shfl_xor(bv, off, 32);
                int   oi = __shfl_xor(bi, off, 32);
                if (ov > bv || (ov == bv && oi < bi)) { bv = ov; bi = oi; }
            }
            if (lane == r) {
                sIdx[wave][r] = bi;
                sY[wave][r]   = bv;
                sT[wave][r]   = ts[ybase + bi];
            }
            if (bi == 2 * lane)     v0 = -INFINITY;
            if (bi == 2 * lane + 1) v1 = -INFINITY;
        }
    }
    __syncthreads();

    // ---- Step 2: gather Mtop (16 rows x 384) -> LDS bf16, float4 loads ----
    {
        const int r   = tid >> 4;        // 0..15
        const int c16 = tid & 15;
        const int t   = r >> 3;
        const long long mrow = ((long long)(b0 + t) * K_TOK + sIdx[t][r & 7]) * D_FEAT;
        const float4* src = (const float4*)(M + mrow);
        #pragma unroll
        for (int j = 0; j < 6; ++j) {
            float4 v = src[c16 + 16 * j];
            int c = (c16 + 16 * j) * 4;
            sA[r][c + 0] = (bf16_t)v.x;
            sA[r][c + 1] = (bf16_t)v.y;
            sA[r][c + 2] = (bf16_t)v.z;
            sA[r][c + 3] = (bf16_t)v.w;
        }
    }
    __syncthreads();

    // ---- Step 3: G = Mtop @ [W1[0:384] | W1[384:768]]  (16x128 f32) ----
    {
        const int nt = wave;                         // N-tile 0..7
        const int colbase = (nt & 3) * 16;           // column in W1
        const int rowoff  = (nt < 4) ? 0 : D_FEAT;   // W1a vs W1b
        const int ar = lane & 15;
        const int kb = (lane < 16) ? 0 : 8;          // A-operand K sub-offset
        v8f acc = {};
        #pragma unroll
        for (int kt = 0; kt < 12; ++kt) {
            v8bf a0 = *(const v8bf*)&sA[ar][kt * 32 + kb];
            v8bf a1 = *(const v8bf*)&sA[ar][kt * 32 + kb + 16];
            v16bf Am = cat8(a0, a1);
            // B operand: lane l holds column l%16, 16 K values in same order as A
            const bf16_t* bp = W1bf + (size_t)(rowoff + kt * 32 + kb) * H_DIM + colbase + ar;
            v16bf Bm;
            #pragma unroll
            for (int s = 0; s < 16; ++s) Bm[s] = bp[(size_t)(s < 8 ? s : s + 8) * H_DIM];
            acc = __builtin_amdgcn_wmma_f32_16x16x32_bf16(
                      false, Am, false, Bm, (short)0, acc, false, false);
        }
        const int col = nt * 16 + (lane & 15);
        #pragma unroll
        for (int v = 0; v < 8; ++v) {
            int row = (lane < 16) ? v : (v + 8);
            sG[row][col] = acc[v];
        }
    }
    if (tid < 64) sPen[tid >> 5][tid & 31] = 0.0f;
    __syncthreads();

    // ---- Step 4: assemble pair pre-activations, silu, store bf16 ----
    for (int item = tid; item < 2 * NPAIR * H_DIM; item += 256) {
        const int n  = item & 63;
        const int pp = item >> 6;
        const int t  = pp / NPAIR;
        const int p  = pp % NPAIR;
        const int i  = c_ii[p], j = c_jj[p];
        const float dt = sT[t][i] - sT[t][j];
        const float yi = sY[t][i], yj = sY[t][j];
        float pre = sG[8 * t + i][n] + sG[8 * t + j][64 + n]
                  + dt * W1[768 * 64 + n]
                  + yi * W1[769 * 64 + n]
                  + yj * W1[770 * 64 + n]
                  + b1[n];
        sH[t][p][n] = (bf16_t)silu_f(pre);
    }
    for (int item = tid; item < 2 * 4 * H_DIM; item += 256) {   // zero pad rows 28..31
        const int n = item & 63;
        const int r = item >> 6;                                 // 0..7
        sH[r >> 2][28 + (r & 3)][n] = (bf16_t)0.0f;
    }
    __syncthreads();

    // ---- Step 5: layer 2 WMMA + silu + dot(W3) + row reduction ----
    {
        const int t   = wave >> 2;        // batch
        const int mt  = (wave >> 1) & 1;  // M-tile (rows 0-15 / 16-31)
        const int nt2 = wave & 1;         // N-tile (cols 0-15 / 16-31)
        const int ar  = mt * 16 + (lane & 15);
        v8f acc = {};
        #pragma unroll
        for (int kt = 0; kt < 2; ++kt) {
            const int kb = kt * 32 + ((lane < 16) ? 0 : 8);
            v8bf a0 = *(const v8bf*)&sH[t][ar][kb];
            v8bf a1 = *(const v8bf*)&sH[t][ar][kb + 16];
            v16bf Am = cat8(a0, a1);
            const bf16_t* bp = W2bf + (size_t)kb * 32 + nt2 * 16 + (lane & 15);
            v16bf Bm;
            #pragma unroll
            for (int s = 0; s < 16; ++s) Bm[s] = bp[(s < 8 ? s : s + 8) * 32];
            acc = __builtin_amdgcn_wmma_f32_16x16x32_bf16(
                      false, Am, false, Bm, (short)0, acc, false, false);
        }
        const int col = nt2 * 16 + (lane & 15);
        const float w3c = W3[col];
        const float b2c = b2[col];
        #pragma unroll
        for (int v = 0; v < 8; ++v) {
            float val = silu_f(acc[v] + b2c) * w3c;
            // reduce the 16 lanes sharing one row (two independent 16-lane groups)
            #pragma unroll
            for (int off = 8; off >= 1; off >>= 1)
                val += __shfl_xor(val, off, 16);
            if ((lane & 15) == 0) {
                int row = mt * 16 + ((lane < 16) ? v : (v + 8));
                atomicAdd(&sPen[t][row], val);   // ds_add_f32
            }
        }
    }
    __syncthreads();

    // ---- Step 6: final weighted sum over 28 pairs ----
    if (wave < 2) {
        const int t = wave;
        float accu = 0.0f;
        if (lane < NPAIR) {
            const float yi = sY[t][c_ii[lane]];
            const float yj = sY[t][c_jj[lane]];
            accu = (sPen[t][lane] + b3[0]) * (yi * yj);
        }
        #pragma unroll
        for (int off = 16; off >= 1; off >>= 1)
            accu += __shfl_xor(accu, off, 32);
        if (lane == 0) out[b0 + t] = accu;
    }
}

// ---------------------------------------------------------------------------
extern "C" void kernel_launch(void* const* d_in, const int* in_sizes, int n_in,
                              void* d_out, int out_size, void* d_ws, size_t ws_size,
                              hipStream_t stream) {
    const float* M   = (const float*)d_in[0];
    const float* y   = (const float*)d_in[1];
    const float* ts  = (const float*)d_in[2];
    const float* W1  = (const float*)d_in[3];
    const float* b1  = (const float*)d_in[4];
    const float* W2  = (const float*)d_in[5];
    const float* b2  = (const float*)d_in[6];
    const float* W3  = (const float*)d_in[7];
    const float* b3  = (const float*)d_in[8];
    float* out = (float*)d_out;

    const int B = in_sizes[1] / K_TOK;   // 8192

    // workspace: bf16 copies of W1 (771x64) and W2 (64x32)
    bf16_t* W1bf = (bf16_t*)d_ws;                       // 98688 bytes
    bf16_t* W2bf = (bf16_t*)((char*)d_ws + 98688);      //  4096 bytes

    pack_weights_bf16<<<(771 * 64 + 255) / 256, 256, 0, stream>>>(W1, W2, W1bf, W2bf);

    recency_energy_kernel<<<B / 2, 256, 0, stream>>>(
        M, y, ts, W1, b1, b2, W3, b3, W1bf, W2bf, out);
}
